// SelectiveNetwork_2929167696579
// MI455X (gfx1250) — matrix operand order repaired and checked by reading, not compiled, not run on hardware
//
#include <hip/hip_runtime.h>
#include <hip/hip_bf16.h>

typedef __bf16 bf16_t;
typedef __attribute__((ext_vector_type(4)))  bf16_t bf16x4;
typedef __attribute__((ext_vector_type(16))) bf16_t bf16x16;
typedef __attribute__((ext_vector_type(4)))  float  f32x4;
typedef __attribute__((ext_vector_type(8)))  float  f32x8;

// async-to-LDS b64 moves 8 bytes: builtin expects pointers to 2xi32 vectors,
// global src in AS1, LDS dst in AS3 (per hipcc diagnostic).
typedef int v2i_t __attribute__((vector_size(8)));
typedef v2i_t __attribute__((address_space(1)))* g64p;
typedef v2i_t __attribute__((address_space(3)))* l64p;

#define B_ROWS 16384
#define C_COLS 1000
#define C_PAD  1024
#define D_DIM  2048

#define BM 128
#define BN 64
#define BK 64
#define LDT 68   // padded LDS tile stride (64 + 4): 136B rows -> conflict-free b64 frag loads

#if __has_builtin(__builtin_amdgcn_global_load_async_to_lds_b64)
#define HAVE_ASYNC_LDS 1
#else
#define HAVE_ASYNC_LDS 0
#endif

__device__ __forceinline__ void wait_asynccnt0() {
#if HAVE_ASYNC_LDS
#if __has_builtin(__builtin_amdgcn_s_wait_asynccnt)
    __builtin_amdgcn_s_wait_asynccnt(0);
#else
    asm volatile("s_wait_asynccnt 0x0" ::: "memory");
#endif
#endif
}

// ---------------------------------------------------------------- conv W -> bf16 (padded)
__global__ __launch_bounds__(256)
void conv_w_kernel(const float* __restrict__ W, bf16_t* __restrict__ wb) {
    size_t idx = ((size_t)blockIdx.x * 256 + threadIdx.x) * 8;   // 8 elems/thread, row-aligned
    int row = (int)(idx / D_DIM);
    bf16x4 t0, t1;
    if (row < C_COLS) {
        f32x4 v0 = *reinterpret_cast<const f32x4*>(W + idx);
        f32x4 v1 = *reinterpret_cast<const f32x4*>(W + idx + 4);
        #pragma unroll
        for (int i = 0; i < 4; ++i) { t0[i] = (bf16_t)v0[i]; t1[i] = (bf16_t)v1[i]; }
    } else {
        #pragma unroll
        for (int i = 0; i < 4; ++i) { t0[i] = (bf16_t)0.0f; t1[i] = (bf16_t)0.0f; }
    }
    *reinterpret_cast<bf16x4*>(wb + idx)     = t0;
    *reinterpret_cast<bf16x4*>(wb + idx + 4) = t1;
}

// ---------------------------------------------------------------- row norms of W
__global__ __launch_bounds__(256)
void w_norms_kernel(const float* __restrict__ W, float* __restrict__ norms) {
    int lane = threadIdx.x & 31;
    int wave = threadIdx.x >> 5;
    int row  = blockIdx.x * 8 + wave;          // grid = 125 -> exactly 1000 rows
    const float* wp = W + (size_t)row * D_DIM;
    float s = 0.0f;
    for (int j = lane; j < D_DIM; j += 32) { float v = wp[j]; s += v * v; }
    #pragma unroll
    for (int off = 16; off; off >>= 1) s += __shfl_xor(s, off, 32);
    if (lane == 0) norms[row] = sqrtf(s);
}

// ---------------------------------------------------------------- GEMM: logits = x * W^T + b
__global__ __launch_bounds__(256)
void gemm_bf16_wmma_kernel(const float* __restrict__ x, const bf16_t* __restrict__ wb,
                           const float* __restrict__ bias, float* __restrict__ out) {
    __shared__ bf16_t xs [2][BM * LDT];
    __shared__ bf16_t wsm[2][BN * LDT];

    const int tid  = threadIdx.x;
    const int lane = tid & 31;
    const int wave = tid >> 5;
    const int wm   = wave & 3;     // wave tile rows: wm*32
    const int wn   = wave >> 2;    // wave tile cols: wn*32

    const int bm0 = blockIdx.y * BM;
    const int bn0 = blockIdx.x * BN;

    f32x8 acc[2][2] = {};

    // staging indices: thread covers rows sr+16p, 4 consecutive elements at col sc
    const int sr = tid >> 4;            // 0..15
    const int sc = (tid & 15) << 2;     // 0,4..60

    // fragment lane decode (ISA 7.12.2, wave32)
    const int lrow = lane & 15;
    const int ka   = (lane < 16) ? 0 : 8;    // A: interleaved 8-chunks {ka..+7, ka+16..+23}
    const int kb   = (lane < 16) ? 0 : 8;    // B: interleaved 8-chunks {kb..+7, kb+16..+23}

    f32x4  xreg[8];
#if !HAVE_ASYNC_LDS
    bf16x4 wreg[4];
#endif

    // issue global loads for tile at k-offset kk; W goes straight to LDS buf nb (async)
    auto stage_global = [&](int kk, int nb) {
        const float* xsrc = x + (size_t)(bm0 + sr) * D_DIM + kk + sc;
        #pragma unroll
        for (int p = 0; p < 8; ++p)
            xreg[p] = *reinterpret_cast<const f32x4*>(xsrc + (size_t)p * 16 * D_DIM);
#if HAVE_ASYNC_LDS
        #pragma unroll
        for (int p = 0; p < 4; ++p)
            __builtin_amdgcn_global_load_async_to_lds_b64(
                (g64p)(wb + (size_t)(bn0 + sr + 16 * p) * D_DIM + kk + sc),
                (l64p)&wsm[nb][(sr + 16 * p) * LDT + sc], 0, 0);
#else
        #pragma unroll
        for (int p = 0; p < 4; ++p)
            wreg[p] = *reinterpret_cast<const bf16x4*>(
                wb + (size_t)(bn0 + sr + 16 * p) * D_DIM + kk + sc);
#endif
        // prefetch x one tile further ahead
        if (kk + BK < D_DIM)
            __builtin_prefetch(xsrc + BK, 0, 0);
    };

    // convert staged x registers and commit to LDS buf nb
    auto stage_commit = [&](int nb) {
        #pragma unroll
        for (int p = 0; p < 8; ++p) {
            bf16x4 t;
            t[0] = (bf16_t)xreg[p][0]; t[1] = (bf16_t)xreg[p][1];
            t[2] = (bf16_t)xreg[p][2]; t[3] = (bf16_t)xreg[p][3];
            *reinterpret_cast<bf16x4*>(&xs[nb][(sr + 16 * p) * LDT + sc]) = t;
        }
#if !HAVE_ASYNC_LDS
        #pragma unroll
        for (int p = 0; p < 4; ++p)
            *reinterpret_cast<bf16x4*>(&wsm[nb][(sr + 16 * p) * LDT + sc]) = wreg[p];
#endif
    };

    // prologue: stage tile 0 into buffer 0
    stage_global(0, 0);
    stage_commit(0);

    const int NK = D_DIM / BK;   // 32
    for (int i = 0; i < NK; ++i) {
        const int cur = i & 1;
        wait_asynccnt0();        // own async W stores into buf[cur] complete
        __syncthreads();         // buf[cur] published; reads of buf[cur^1] done -> safe to overwrite

        if (i + 1 < NK) stage_global((i + 1) * BK, cur ^ 1);   // latency overlapped with compute

        // ---- compute from buf[cur]: 2 K-substeps of 32, 4 WMMAs each ----
        #pragma unroll
        for (int ks = 0; ks < BK; ks += 32) {
            bf16x16 Af[2], Bf[2];
            #pragma unroll
            for (int mt = 0; mt < 2; ++mt) {
                const bf16x4* p4 = reinterpret_cast<const bf16x4*>(
                    &xs[cur][(wm * 32 + mt * 16 + lrow) * LDT + ks + ka]);
                bf16x4 c0 = p4[0], c1 = p4[1], c2 = p4[4], c3 = p4[5];
                #pragma unroll
                for (int q = 0; q < 4; ++q) {
                    Af[mt][q]      = c0[q];
                    Af[mt][4 + q]  = c1[q];
                    Af[mt][8 + q]  = c2[q];
                    Af[mt][12 + q] = c3[q];
                }
            }
            #pragma unroll
            for (int nt = 0; nt < 2; ++nt) {
                const bf16x4* p4 = reinterpret_cast<const bf16x4*>(
                    &wsm[cur][(wn * 32 + nt * 16 + lrow) * LDT + ks + kb]);
                #pragma unroll
                for (int c = 0; c < 4; ++c) {
                    bf16x4 cc = p4[(c & 1) + (c >> 1) * 4];
                    #pragma unroll
                    for (int q = 0; q < 4; ++q) Bf[nt][c * 4 + q] = cc[q];
                }
            }
            #pragma unroll
            for (int mt = 0; mt < 2; ++mt)
                #pragma unroll
                for (int nt = 0; nt < 2; ++nt)
                    acc[mt][nt] = __builtin_amdgcn_wmma_f32_16x16x32_bf16(
                        false, Af[mt], false, Bf[nt], (short)0, acc[mt][nt], false, false);
        }

        if (i + 1 < NK) stage_commit(cur ^ 1);   // loadcnt wait lands here, after the WMMAs
    }

    // ---- epilogue: C/D layout -> lane N = lane%16, VGPR i -> M = i (+8 for upper half-wave)
    const int ncol = lane & 15;
    const int mofs = (lane < 16) ? 0 : 8;
    #pragma unroll
    for (int nt = 0; nt < 2; ++nt) {
        int n = bn0 + wn * 32 + nt * 16 + ncol;
        if (n < C_COLS) {
            float bv = bias[n];
            #pragma unroll
            for (int mt = 0; mt < 2; ++mt) {
                int m0 = bm0 + wm * 32 + mt * 16 + mofs;
                #pragma unroll
                for (int q = 0; q < 8; ++q)
                    out[(size_t)(m0 + q) * C_COLS + n] = acc[mt][nt][q] + bv;
            }
        }
    }
}

// ---------------------------------------------------------------- selector scores per row
__device__ __forceinline__ void merge_top2(float& m1, float& m2, float o1, float o2) {
    float hi = fmaxf(m1, o1);
    float lo = fmaxf(fminf(m1, o1), fmaxf(m2, o2));
    m1 = hi; m2 = lo;
}

__global__ __launch_bounds__(256)
void selectors_kernel(const float* __restrict__ logits, const float* __restrict__ norms,
                      const float* __restrict__ selw, const float* __restrict__ selt,
                      const float* __restrict__ ethr, float* __restrict__ outsel) {
    const int row  = blockIdx.x;
    const int tid  = threadIdx.x;
    const int lane = tid & 31;
    const int wave = tid >> 5;
    const float NEG = -__builtin_inff();
    const float* lp = logits + (size_t)row * C_COLS;

    float v[4], g[4];
    #pragma unroll
    for (int j = 0; j < 4; ++j) {
        int c = tid + j * 256;
        if (c < C_COLS) { v[j] = lp[c]; g[j] = v[j] / (norms[c] + 1e-8f); }
        else            { v[j] = NEG;   g[j] = NEG; }
    }

    // pass 1: top-2 of raw logits and of normalized logits
    float m1 = NEG, m2 = NEG, g1 = NEG, g2 = NEG;
    #pragma unroll
    for (int j = 0; j < 4; ++j) {
        if (v[j] > m1) { m2 = m1; m1 = v[j]; } else if (v[j] > m2) m2 = v[j];
        if (g[j] > g1) { g2 = g1; g1 = g[j]; } else if (g[j] > g2) g2 = g[j];
    }
    #pragma unroll
    for (int off = 16; off; off >>= 1) {
        float o1 = __shfl_xor(m1, off, 32), o2 = __shfl_xor(m2, off, 32);
        merge_top2(m1, m2, o1, o2);
        float p1 = __shfl_xor(g1, off, 32), p2 = __shfl_xor(g2, off, 32);
        merge_top2(g1, g2, p1, p2);
    }

    __shared__ float red[8][4];
    __shared__ float bres[4];
    if (lane == 0) { red[wave][0] = m1; red[wave][1] = m2; red[wave][2] = g1; red[wave][3] = g2; }
    __syncthreads();
    if (tid == 0) {
        for (int w = 1; w < 8; ++w) {
            merge_top2(m1, m2, red[w][0], red[w][1]);
            merge_top2(g1, g2, red[w][2], red[w][3]);
        }
        bres[0] = m1; bres[1] = m2; bres[2] = g1; bres[3] = g2;
    }
    __syncthreads();
    const float rmax = bres[0];

    // pass 2: softmax sums with d = l - max
    float s1 = 0.0f, s2 = 0.0f, s3 = 0.0f;
    #pragma unroll
    for (int j = 0; j < 4; ++j) {
        int c = tid + j * 256;
        if (c < C_COLS) {
            float d = v[j] - rmax;
            float e = __expf(d);
            s1 += e; s2 += e * e; s3 += d * e;
        }
    }
    #pragma unroll
    for (int off = 16; off; off >>= 1) {
        s1 += __shfl_xor(s1, off, 32);
        s2 += __shfl_xor(s2, off, 32);
        s3 += __shfl_xor(s3, off, 32);
    }
    if (lane == 0) { red[wave][0] = s1; red[wave][1] = s2; red[wave][2] = s3; }
    __syncthreads();
    if (tid == 0) {
        for (int w = 1; w < 8; ++w) { s1 += red[w][0]; s2 += red[w][1]; s3 += red[w][2]; }
        float tm1 = bres[0], tm2 = bres[1], tg1 = bres[2], tg2 = bres[3];
        float sr_max = 1.0f / s1;                  // exp(0)/sum
        float doctor = 1.0f - (s1 * s1) / s2;      // 1 - 1/sum(p^2)
        float negent = s3 / s1 - __logf(s1);       // sum p*log p
        float sc[5] = { sr_max, doctor, negent, tg1 - tg2, tm1 - tm2 };
        float ens = 0.0f;
        #pragma unroll
        for (int i = 0; i < 5; ++i) ens += selw[i] * tanhf(sc[i] - selt[i]);
        outsel[row] = tanhf(ens - ethr[0]);
    }
}

// ---------------------------------------------------------------- launch
extern "C" void kernel_launch(void* const* d_in, const int* in_sizes, int n_in,
                              void* d_out, int out_size, void* d_ws, size_t ws_size,
                              hipStream_t stream) {
    const float* x    = (const float*)d_in[0];
    const float* W    = (const float*)d_in[1];
    const float* bias = (const float*)d_in[2];
    const float* selw = (const float*)d_in[3];
    const float* selt = (const float*)d_in[4];
    const float* ethr = (const float*)d_in[5];

    float*  out   = (float*)d_out;
    bf16_t* wb    = (bf16_t*)d_ws;                                   // 1024*2048 bf16
    float*  norms = (float*)((char*)d_ws + (size_t)C_PAD * D_DIM * sizeof(bf16_t));

    conv_w_kernel   <<<(C_PAD * D_DIM) / (256 * 8), 256, 0, stream>>>(W, wb);
    w_norms_kernel  <<<C_COLS / 8,                  256, 0, stream>>>(W, norms);
    gemm_bf16_wmma_kernel<<<dim3(C_PAD / BN, B_ROWS / BM), 256, 0, stream>>>(x, wb, bias, out);
    selectors_kernel<<<B_ROWS, 256, 0, stream>>>(out, norms, selw, selt, ethr,
                                                 out + (size_t)B_ROWS * C_COLS);
}
